// SelfAttention_26096221291158
// MI455X (gfx1250) — matrix operand order repaired and checked by reading, not compiled, not run on hardware
//
#include <hip/hip_runtime.h>
#include <hip/hip_bf16.h>
#include <cstdint>

// ---------------------------------------------------------------------------
// Self-attention with softmax over axis=1 (column-wise), bf16 WMMA pipeline.
//   B=8, N=2048, d=512.
//   S = Q K^T * 1/sqrt(512)                      (v_wmma_f32_16x16x32_bf16)
//   att[n,m] = exp(S[n,m]-max_n) / sum_n(...)    (column softmax; fold 1/sum into V)
//   out = att @ V                                (v_wmma_f32_16x16x32_bf16)
// Wave tile: 64(M) x 32(N)  -> 8 WMMA / 12 b128-loads per k-step.
// Block: 8 waves = 2(M) x 4(N) -> 128x128 block tile.
// k-loop unrolled x2 so the scheduler overlaps next-step loads with WMMAs.
// ---------------------------------------------------------------------------

typedef __bf16 bf16_t;
typedef __attribute__((ext_vector_type(16))) __bf16 v16bf;
typedef __attribute__((ext_vector_type(8)))  __bf16 v8bf;
typedef __attribute__((ext_vector_type(4)))  __bf16 v4bf;
typedef __attribute__((ext_vector_type(8)))  float  v8f;

#define BB   8
#define NN   2048
#define DD   512
#define BN   (BB * NN)          // 16384 tokens total
#define SCALE 0.044194173824159216f   // 1/sqrt(512)

// ---- WMMA helpers ----------------------------------------------------------
static __device__ __forceinline__ v8f wmma_bf16(v16bf a, v16bf b, v8f c) {
  return __builtin_amdgcn_wmma_f32_16x16x32_bf16(
      /*neg_a=*/false, a, /*neg_b=*/false, b, /*c_mod=*/(short)0, c,
      /*reuse_a=*/false, /*reuse_b=*/false);
}

// A-frag: lane (m = lane&15, h = lane>>4) holds K = {8h..8h+7} U {16+8h..16+8h+7}
// => two contiguous 16B chunks from a K-contiguous (row-major) source row.
static __device__ __forceinline__ v16bf load_a_frag(const bf16_t* rowp, int h) {
  v8bf lo = *(const v8bf*)(rowp + 8 * h);
  v8bf hi = *(const v8bf*)(rowp + 8 * h + 16);
  return __builtin_shufflevector(lo, hi, 0,1,2,3,4,5,6,7,8,9,10,11,12,13,14,15);
}

// B-frag: lane (n = lane&15, h = lane>>4) holds K = {8h..8h+7} U {16+8h..16+8h+7}
// => two contiguous 16B chunks from a K-contiguous source row (the "column" n).
static __device__ __forceinline__ v16bf load_b_frag(const bf16_t* colp, int h) {
  v8bf lo = *(const v8bf*)(colp + 8 * h);
  v8bf hi = *(const v8bf*)(colp + 8 * h + 16);
  return __builtin_shufflevector(lo, hi, 0,1,2,3,4,5,6,7,8,9,10,11,12,13,14,15);
}

static __device__ __forceinline__ v8f zero8() {
  v8f z;
#pragma unroll
  for (int i = 0; i < 8; ++i) z[i] = 0.f;
  return z;
}

// ---- shared 4x2-subtile WMMA inner loop ------------------------------------
// acc[i][j] accumulates the 16x16 tile at rows m0+16i, cols n0+16j.
static __device__ __forceinline__ void mma_tile_loop(
    const bf16_t* __restrict__ A, size_t lda,
    const bf16_t* __restrict__ Bm, size_t ldb,
    int Kdim, int m0, int n0, int l15, int h, v8f acc[4][2]) {
  const bf16_t* ap[4];
  const bf16_t* bp[2];
#pragma unroll
  for (int i = 0; i < 4; ++i) ap[i] = A + (size_t)(m0 + 16 * i + l15) * lda;
#pragma unroll
  for (int j = 0; j < 2; ++j) bp[j] = Bm + (size_t)(n0 + 16 * j + l15) * ldb;

#pragma unroll 2
  for (int k = 0; k < Kdim; k += 32) {
    v16bf af[4], bfv[2];
#pragma unroll
    for (int i = 0; i < 4; ++i) af[i] = load_a_frag(ap[i] + k, h);
#pragma unroll
    for (int j = 0; j < 2; ++j) bfv[j] = load_b_frag(bp[j] + k, h);
#pragma unroll
    for (int i = 0; i < 4; ++i)
#pragma unroll
      for (int j = 0; j < 2; ++j)
        acc[i][j] = wmma_bf16(af[i], bfv[j], acc[i][j]);
  }
}

// ---- prep: x f32 -> bf16 (4 elems/thread) ----------------------------------
__global__ __launch_bounds__(256) void cvt_x(const float* __restrict__ x,
                                             bf16_t* __restrict__ xb) {
  int i = blockIdx.x * 256 + threadIdx.x;           // 0 .. BN*DD/4-1
  const float4 v = ((const float4*)x)[i];
  v4bf o;
  o[0] = (bf16_t)v.x; o[1] = (bf16_t)v.y; o[2] = (bf16_t)v.z; o[3] = (bf16_t)v.w;
  ((v4bf*)xb)[i] = o;
}

// ---- prep: transpose + convert the three 512x512 weights -------------------
__global__ __launch_bounds__(256) void prep_w(const float* __restrict__ Wq,
                                              const float* __restrict__ Wk,
                                              const float* __restrict__ Wv,
                                              bf16_t* __restrict__ Wqt,
                                              bf16_t* __restrict__ Wkt,
                                              bf16_t* __restrict__ Wvt) {
  const int which = blockIdx.y;
  const float* W = (which == 0) ? Wq : ((which == 1) ? Wk : Wv);
  bf16_t* Wt     = (which == 0) ? Wqt : ((which == 1) ? Wkt : Wvt);
  int idx = blockIdx.x * 256 + threadIdx.x;         // 0 .. 512*512-1
  int o = idx >> 9, k = idx & 511;
  Wt[idx] = (bf16_t)W[(size_t)k * DD + o];          // Wt[o][k] = W[k][o]
}

// ---- generic 512-deep projection GEMM: C = A(Mx512) * Bm^T-layout + bias ---
// A row-major [M,512] bf16; Bm rows are output columns [N,512] bf16 (K-contig).
// bias_per_row==0: bias indexed by output column n (Q,K projections)
// bias_per_row==1: bias indexed by output row m (V^T projection)
__global__ __launch_bounds__(256) void proj_gemm(const bf16_t* __restrict__ A,
                                                 const bf16_t* __restrict__ Bm,
                                                 const float*  __restrict__ bias,
                                                 bf16_t* __restrict__ C,
                                                 int ldc, int bias_per_row) {
  const int lane = threadIdx.x & 31;
  const int wave = threadIdx.x >> 5;
  const int h    = lane >> 4;
  const int l15  = lane & 15;
  const int m0 = blockIdx.y * 128 + (wave & 1) * 64;   // 2 waves down M
  const int n0 = blockIdx.x * 128 + (wave >> 1) * 32;  // 4 waves across N

  v8f acc[4][2];
#pragma unroll
  for (int i = 0; i < 4; ++i)
#pragma unroll
    for (int j = 0; j < 2; ++j) acc[i][j] = zero8();

  mma_tile_loop(A, DD, Bm, DD, DD, m0, n0, l15, h, acc);

#pragma unroll
  for (int i = 0; i < 4; ++i)
#pragma unroll
    for (int j = 0; j < 2; ++j) {
      const int mb = m0 + 16 * i + 8 * h;
      const int n  = n0 + 16 * j + l15;
#pragma unroll
      for (int r = 0; r < 8; ++r) {
        const int m = mb + r;
        float v = acc[i][j][r] + (bias_per_row ? bias[m] : bias[n]);
        C[(size_t)m * ldc + n] = (bf16_t)v;
      }
    }
}

// ---- scores: S[b][n][m] = (Q[b,n,:] . K[b,m,:]) * SCALE --------------------
__global__ __launch_bounds__(256) void scores_gemm(const bf16_t* __restrict__ Q,
                                                   const bf16_t* __restrict__ K,
                                                   float* __restrict__ S) {
  const int lane = threadIdx.x & 31;
  const int wave = threadIdx.x >> 5;
  const int h    = lane >> 4;
  const int l15  = lane & 15;
  const int b  = blockIdx.z;
  const int m0 = blockIdx.y * 128 + (wave & 1) * 64;   // query rows n
  const int n0 = blockIdx.x * 128 + (wave >> 1) * 32;  // key cols m

  const bf16_t* Qb = Q + (size_t)b * NN * DD;
  const bf16_t* Kb = K + (size_t)b * NN * DD;
  float*        Sb = S + (size_t)b * NN * NN;

  v8f acc[4][2];
#pragma unroll
  for (int i = 0; i < 4; ++i)
#pragma unroll
    for (int j = 0; j < 2; ++j) acc[i][j] = zero8();

  mma_tile_loop(Qb, DD, Kb, DD, DD, m0, n0, l15, h, acc);

#pragma unroll
  for (int i = 0; i < 4; ++i)
#pragma unroll
    for (int j = 0; j < 2; ++j) {
      const int mb = m0 + 16 * i + 8 * h;
      const int n  = n0 + 16 * j + l15;
#pragma unroll
      for (int r = 0; r < 8; ++r)
        Sb[(size_t)(mb + r) * NN + n] = acc[i][j][r] * SCALE;
    }
}

// ---- column softmax (axis=1): one thread per column m; coalesced rows ------
// Writes UNNORMALIZED exp(S - colmax) as bf16; colsum saved for folding into V.
__global__ __launch_bounds__(256) void col_softmax(const float* __restrict__ S,
                                                   bf16_t* __restrict__ attE,
                                                   float* __restrict__ colsum) {
  const int b = blockIdx.y;
  const int m = blockIdx.x * 256 + threadIdx.x;
  const float* Sb = S    + (size_t)b * NN * NN + m;
  bf16_t*      Ab = attE + (size_t)b * NN * NN + m;
  float mx = -3.402823466e38f;
  for (int n = 0; n < NN; ++n) mx = fmaxf(mx, Sb[(size_t)n * NN]);
  float s = 0.f;
  for (int n = 0; n < NN; ++n) {
    float e = __expf(Sb[(size_t)n * NN] - mx);
    s += e;
    Ab[(size_t)n * NN] = (bf16_t)e;
  }
  colsum[b * NN + m] = s;
}

// ---- fold 1/colsum into V^T rows (in place; V^T is rebuilt every call) -----
__global__ __launch_bounds__(256) void scale_v(bf16_t* __restrict__ Vt,
                                               const float* __restrict__ colsum) {
  const int g = blockIdx.x * 256 + threadIdx.x;   // global token 0..16383
  const int f = blockIdx.y;                       // feature 0..511
  const float r = 1.0f / colsum[g];
  const size_t idx = (size_t)f * BN + g;
  Vt[idx] = (bf16_t)((float)Vt[idx] * r);
}

// ---- out[b][n][v] = sum_m attE[b][n][m] * Vt_scaled[v][b*NN+m] -------------
__global__ __launch_bounds__(256) void out_gemm(const bf16_t* __restrict__ attE,
                                                const bf16_t* __restrict__ Vt,
                                                float* __restrict__ Out) {
  const int lane = threadIdx.x & 31;
  const int wave = threadIdx.x >> 5;
  const int h    = lane >> 4;
  const int l15  = lane & 15;
  const int b  = blockIdx.z;
  const int m0 = blockIdx.y * 128 + (wave & 1) * 64;   // output rows n
  const int n0 = blockIdx.x * 128 + (wave >> 1) * 32;  // output cols v

  const bf16_t* Ab = attE + (size_t)b * NN * NN;       // [n][m], m contiguous
  const bf16_t* Vb = Vt   + (size_t)b * NN;            // [v][token], ld=BN
  float*        Ob = Out  + (size_t)b * NN * DD;

  v8f acc[4][2];
#pragma unroll
  for (int i = 0; i < 4; ++i)
#pragma unroll
    for (int j = 0; j < 2; ++j) acc[i][j] = zero8();

  mma_tile_loop(Ab, NN, Vb, BN, NN, m0, n0, l15, h, acc);

#pragma unroll
  for (int i = 0; i < 4; ++i)
#pragma unroll
    for (int j = 0; j < 2; ++j) {
      const int mb = m0 + 16 * i + 8 * h;
      const int n  = n0 + 16 * j + l15;
#pragma unroll
      for (int r = 0; r < 8; ++r)
        Ob[(size_t)(mb + r) * DD + n] = acc[i][j][r];
    }
}

// ---------------------------------------------------------------------------
extern "C" void kernel_launch(void* const* d_in, const int* in_sizes, int n_in,
                              void* d_out, int out_size, void* d_ws, size_t ws_size,
                              hipStream_t stream) {
  (void)in_sizes; (void)n_in; (void)out_size; (void)ws_size;

  const float* x  = (const float*)d_in[0];
  const float* Wq = (const float*)d_in[1];
  const float* bq = (const float*)d_in[2];
  const float* Wk = (const float*)d_in[3];
  const float* bk = (const float*)d_in[4];
  const float* Wv = (const float*)d_in[5];
  const float* bv = (const float*)d_in[6];
  float* out = (float*)d_out;

  // workspace layout (bytes)
  char* ws = (char*)d_ws;
  bf16_t* x_bf = (bf16_t*)(ws);                          // 16 MB  [BN, DD]
  bf16_t* Wqt  = (bf16_t*)(ws + 16777216);               // 512KB  [DD, DD] (transposed)
  bf16_t* Wkt  = (bf16_t*)(ws + 17301504);               // 512KB
  bf16_t* Wvt  = (bf16_t*)(ws + 17825792);               // 512KB
  bf16_t* Qb   = (bf16_t*)(ws + 18350080);               // 16 MB  [BN, DD]
  bf16_t* Kb   = (bf16_t*)(ws + 35127296);               // 16 MB  [BN, DD]
  bf16_t* Vt   = (bf16_t*)(ws + 51904512);               // 16 MB  [DD, BN] (V transposed)
  float*  S    = (float*) (ws + 68681728);               // 128 MB [B, NN, NN]
  bf16_t* attE = (bf16_t*)(ws + 202899456);              // 64 MB  [B, NN, NN] unnormalized exp
  float*  csum = (float*) (ws + 270008320);              // 64 KB  [B, NN]

  // 1) precision staging
  cvt_x <<<BN * DD / (256 * 4), 256, 0, stream>>>(x, x_bf);
  prep_w<<<dim3(DD * DD / 256, 3), 256, 0, stream>>>(Wq, Wk, Wv, Wqt, Wkt, Wvt);

  // 2) projections (bf16 WMMA, f32 accum, fused bias); block tile 128x128
  proj_gemm<<<dim3(DD / 128, BN / 128), 256, 0, stream>>>(x_bf, Wqt, bq, Qb, DD, 0);
  proj_gemm<<<dim3(DD / 128, BN / 128), 256, 0, stream>>>(x_bf, Wkt, bk, Kb, DD, 0);
  // V computed directly transposed: Vt[v][token] = (x @ Wv + bv)^T
  proj_gemm<<<dim3(BN / 128, DD / 128), 256, 0, stream>>>(Wvt, x_bf, bv, Vt, BN, 1);

  // 3) scores S = Q K^T * scale, per batch (L2-resident: 16.8 MB/batch)
  scores_gemm<<<dim3(NN / 128, NN / 128, BB), 256, 0, stream>>>(Qb, Kb, S);

  // 4) softmax over axis=1 (columns); store unnormalized exp, keep colsum
  col_softmax<<<dim3(NN / 256, BB), 256, 0, stream>>>(S, attE, csum);

  // 5) fold 1/colsum into V^T (16 MB instead of renormalizing 64 MB of att)
  scale_v<<<dim3(BN / 256, DD), 256, 0, stream>>>(Vt, csum);

  // 6) out = attE @ Vs
  out_gemm<<<dim3(DD / 128, NN / 128, BB), 256, 0, stream>>>(attE, Vt, out);
}
